// Seq2Seq_pointer_generator_20194936226683
// MI455X (gfx1250) — matrix operand order repaired and checked by reading, not compiled, not run on hardware
//
#include <hip/hip_runtime.h>
#include <math.h>

// ---------------- problem constants ----------------
constexpr int V   = 50000;
constexpr int E   = 128;
constexpr int H   = 256;
constexpr int B   = 32;
constexpr int S   = 400;
constexpr int T   = 16;
constexpr int OOV = 20;
constexpr int H2  = 2 * H;        // 512
constexpr int H3  = 3 * H;        // 768
constexpr int EH2 = E + 2 * H;    // 640
constexpr int EH3 = E + 3 * H;    // 896
constexpr int VE  = V + OOV;      // 50020
constexpr int SB  = S * B;        // 12800

typedef _Float16 half_t;
typedef __attribute__((ext_vector_type(8)))  _Float16 v8h;
typedef __attribute__((ext_vector_type(16))) _Float16 v16h;
typedef __attribute__((ext_vector_type(8)))  float    v8f;

// ---------------- WMMA tile helper (double-buffered) ----------------
// Accumulate one 16x16 f32 tile of D += A(16,K) * W(16,K)^T using
// v_wmma_f32_16x16x32_f16. A rows contiguous with stride lda, W rows
// (output columns) contiguous with stride ldw. K % 32 == 0.
// Per-lane operand layout used here (contiguous k, same for A and B):
//   A: row = lane&15, K elems kbase..kbase+15,              kbase = (lane>=16)*16
//   B: col = lane&15, K elems kbase..kbase+15,              kbase = (lane>=16)*16
// All call sites guarantee 32-byte alignment of the vector loads
// (buffers 256-B aligned, row strides multiples of 32 halves).
__device__ __forceinline__ void wmma_load_ab(const half_t* ap, const half_t* bp,
                                             int kt, v16h& a, v16h& b) {
  a = *(const v16h*)(ap + kt);
  b = *(const v16h*)(bp + kt);
}

__device__ __forceinline__ v8f wmma_k_f16(const half_t* A, int lda,
                                          const half_t* W, int ldw,
                                          int K, v8f acc)
{
  const int lane = threadIdx.x & 31;
  const int r    = lane & 15;
  const int hi   = lane >> 4;
  const half_t* ap = A + r * lda + hi * 16;
  const half_t* bp = W + r * ldw + hi * 16;
  v16h a0, b0;
  wmma_load_ab(ap, bp, 0, a0, b0);               // prologue: slice 0 in flight
  for (int kt = 32; kt < K; kt += 32) {
    v16h a1, b1;
    wmma_load_ab(ap, bp, kt, a1, b1);            // issue next slice's loads
    __builtin_prefetch((const void*)(bp + kt + 32), 0, 1);  // global_prefetch_b8
    acc = __builtin_amdgcn_wmma_f32_16x16x32_f16(
        /*neg_a=*/false, a0, /*neg_b=*/false, b0,
        /*c_mod=*/(short)0, acc, /*reuse_a=*/false, /*reuse_b=*/false);
    a0 = a1; b0 = b1;                            // rotate double buffer
  }
  acc = __builtin_amdgcn_wmma_f32_16x16x32_f16(
      false, a0, false, b0, (short)0, acc, false, false);
  return acc;
}

__device__ __forceinline__ float sigf(float x) { return 1.0f / (1.0f + __expf(-x)); }

__device__ __forceinline__ float block_reduce_sum(float v, float* red) {
  const int tid = threadIdx.x;
  red[tid] = v; __syncthreads();
  for (int o = 128; o > 0; o >>= 1) {
    if (tid < o) red[tid] += red[tid + o];
    __syncthreads();
  }
  float r = red[0]; __syncthreads();
  return r;
}
__device__ __forceinline__ float block_reduce_max(float v, float* red) {
  const int tid = threadIdx.x;
  red[tid] = v; __syncthreads();
  for (int o = 128; o > 0; o >>= 1) {
    if (tid < o) red[tid] = fmaxf(red[tid], red[tid + o]);
    __syncthreads();
  }
  float r = red[0]; __syncthreads();
  return r;
}

// ---------------- small utility kernels ----------------
__global__ void k_cast_f32_f16(const float* __restrict__ src, half_t* __restrict__ dst, int n) {
  int i = blockIdx.x * 256 + threadIdx.x;
  if (i < n) dst[i] = (half_t)src[i];
}

__global__ void k_zero(float* p, int n) {
  int i = blockIdx.x * 256 + threadIdx.x;
  if (i < n) p[i] = 0.0f;
}

// Gather source embeddings -> f16, layout (S*B, E), row = s*B + b.
__global__ void k_embed_src(const int* __restrict__ input_seq,
                            const float* __restrict__ embedding,
                            half_t* __restrict__ embH) {
  int i = blockIdx.x * 256 + threadIdx.x;
  if (i >= SB * E) return;
  int row = i >> 7;        // /E
  int k   = i & (E - 1);
  int s   = row / B;
  int b   = row % B;
  int tok = input_seq[b * S + s];
  embH[i] = (half_t)embedding[(size_t)tok * E + k];
}

// Gather decoder-input token embeddings into xin[:,0:E] and cat[:,0:E].
__global__ void k_embed_tok(const int* __restrict__ target_seq,
                            const float* __restrict__ embedding,
                            half_t* __restrict__ xinH, half_t* __restrict__ catH,
                            int t0) {
  int i = blockIdx.x * 256 + threadIdx.x;
  if (i >= B * E) return;
  int b = i >> 7;
  int k = i & (E - 1);
  int tok = target_seq[b * T + t0];
  half_t v = (half_t)embedding[(size_t)tok * E + k];
  xinH[b * EH2 + k] = v;
  catH[b * EH3 + k] = v;
}

// ---------------- generic WMMA GEMM ----------------
// C(M,N) = A(M,K) * W(N,K)^T + bias ; optional tanh ; optional f16 copy.
__global__ __launch_bounds__(256) void k_gemm(const half_t* __restrict__ A,
                                              const half_t* __restrict__ W,
                                              const float* __restrict__ bias,
                                              float* __restrict__ C,
                                              half_t* __restrict__ C16,
                                              int M, int N, int K, int act) {
  int gw    = (blockIdx.x * blockDim.x + threadIdx.x) >> 5;
  int tn    = N >> 4;
  int total = (M >> 4) * tn;
  if (gw >= total) return;
  int mt = gw / tn, nt = gw % tn;
  const int lane = threadIdx.x & 31;
  const int r    = lane & 15;
  const int hi   = lane >> 4;
  v8f acc = {};
  acc = wmma_k_f16(A + (size_t)mt * 16 * K, K, W + (size_t)nt * 16 * K, K, K, acc);
  int n = nt * 16 + r;
  float bv = bias ? bias[n] : 0.0f;
#pragma unroll
  for (int v = 0; v < 8; ++v) {
    int m = mt * 16 + hi * 8 + v;
    float val = acc[v] + bv;
    if (act == 1) val = tanhf(val);
    C[(size_t)m * N + n] = val;
    if (C16) C16[(size_t)m * N + n] = (half_t)val;
  }
}

// ---------------- persistent bidirectional GRU encoder ----------------
// One block per direction (blockIdx.x: 0=fwd, 1=bwd). h lives in LDS.
__global__ __launch_bounds__(256) void k_encoder(
    const half_t* __restrict__ embH,
    const half_t* __restrict__ WihF, const half_t* __restrict__ WhhF,
    const float* __restrict__ bihF,  const float* __restrict__ bhhF,
    const half_t* __restrict__ WihB, const half_t* __restrict__ WhhB,
    const float* __restrict__ bihB,  const float* __restrict__ bhhB,
    half_t* __restrict__ encH, float* __restrict__ encF,
    half_t* __restrict__ hTh)
{
  const int dir = blockIdx.x;
  const half_t* Wih = dir ? WihB : WihF;
  const half_t* Whh = dir ? WhhB : WhhF;
  const float*  bih = dir ? bihB : bihF;
  const float*  bhh = dir ? bhhB : bhhF;

  __shared__ float  hF[B][H];
  __shared__ half_t hH[B][H];
  for (int i = threadIdx.x; i < B * H; i += 256) {
    (&hF[0][0])[i] = 0.0f;
    (&hH[0][0])[i] = (half_t)0.0f;
  }
  __syncthreads();

  const int lane = threadIdx.x & 31;
  const int wave = threadIdx.x >> 5;
  const int r    = lane & 15;
  const int hi   = lane >> 4;

  for (int t = 0; t < S; ++t) {
    const int ts = dir ? (S - 1 - t) : t;
    const half_t* xbase = embH + (size_t)ts * B * E;
    float hnew[4][8];
    // compute phase: all 32 (mt,j) tile-groups, 4 per wave
    for (int g4 = 0; g4 < 4; ++g4) {
      int g = wave + g4 * 8;
      int mt = g >> 4, j = g & 15;
      const half_t* xrow = xbase + mt * 16 * E;
      const half_t* hrow = &hH[mt * 16][0];
      v8f i0 = {}, i1 = {}, i2 = {}, h0 = {}, h1 = {}, h2 = {};
      i0 = wmma_k_f16(xrow, E, Wih + (size_t)(0 * H + j * 16) * E, E, E, i0);
      i1 = wmma_k_f16(xrow, E, Wih + (size_t)(1 * H + j * 16) * E, E, E, i1);
      i2 = wmma_k_f16(xrow, E, Wih + (size_t)(2 * H + j * 16) * E, E, E, i2);
      h0 = wmma_k_f16(hrow, H, Whh + (size_t)(0 * H + j * 16) * H, H, H, h0);
      h1 = wmma_k_f16(hrow, H, Whh + (size_t)(1 * H + j * 16) * H, H, H, h1);
      h2 = wmma_k_f16(hrow, H, Whh + (size_t)(2 * H + j * 16) * H, H, H, h2);
      int col = j * 16 + r;
      float br = bih[col], bz = bih[H + col], bn = bih[2 * H + col];
      float cr = bhh[col], cz = bhh[H + col], cn = bhh[2 * H + col];
#pragma unroll
      for (int v = 0; v < 8; ++v) {
        int m = mt * 16 + hi * 8 + v;
        float rr = sigf((i0[v] + br) + (h0[v] + cr));
        float zz = sigf((i1[v] + bz) + (h1[v] + cz));
        float nn = tanhf((i2[v] + bn) + rr * (h2[v] + cn));
        float ho = hF[m][col];
        hnew[g4][v] = (1.0f - zz) * nn + zz * ho;
      }
    }
    __syncthreads();
    // write phase: update h, stream enc_states (f32 + f16)
    for (int g4 = 0; g4 < 4; ++g4) {
      int g = wave + g4 * 8;
      int mt = g >> 4, j = g & 15;
      int col = j * 16 + r;
#pragma unroll
      for (int v = 0; v < 8; ++v) {
        int m = mt * 16 + hi * 8 + v;
        float hv = hnew[g4][v];
        half_t hh = (half_t)hv;
        hF[m][col] = hv;
        hH[m][col] = hh;
        size_t idx = ((size_t)ts * B + m) * H2 + dir * H + col;
        encF[idx] = hv;
        encH[idx] = hh;
      }
    }
    __syncthreads();
  }
  // final hidden -> bridge input (B, 2H) f16
  for (int i = threadIdx.x; i < B * H; i += 256) {
    int m = i / H, c = i % H;
    hTh[m * H2 + dir * H + c] = hH[m][c];
  }
}

// ---------------- attention (block per batch row) ----------------
__global__ __launch_bounds__(256) void k_attn(
    const float* __restrict__ q, const float* __restrict__ proj,
    const float* __restrict__ We, const float* __restrict__ encF,
    float* __restrict__ alphas, half_t* __restrict__ xinH,
    half_t* __restrict__ catH)
{
  const int b = blockIdx.x;
  __shared__ float sc[S];
  __shared__ float qs[H];
  __shared__ float red[256];
  for (int i = threadIdx.x; i < H; i += 256) qs[i] = q[b * H + i];
  __syncthreads();
  for (int s = threadIdx.x; s < S; s += 256) {
    const float* pr = proj + ((size_t)s * B + b) * H;
    float acc = 0.0f;
    for (int d = 0; d < H; ++d) acc += tanhf(qs[d] + pr[d]) * We[d];
    sc[s] = acc;
  }
  __syncthreads();
  float mx = -3.0e38f;
  for (int s = threadIdx.x; s < S; s += 256) mx = fmaxf(mx, sc[s]);
  mx = block_reduce_max(mx, red);
  float sum = 0.0f;
  for (int s = threadIdx.x; s < S; s += 256) {
    float e = __expf(sc[s] - mx);
    sc[s] = e;
    sum += e;
  }
  sum = block_reduce_sum(sum, red);
  float inv = 1.0f / sum;
  for (int s = threadIdx.x; s < S; s += 256) {
    float a = sc[s] * inv;
    sc[s] = a;
    alphas[(size_t)s * B + b] = a;
  }
  __syncthreads();
  // context over 2H dims
  for (int d = threadIdx.x; d < H2; d += 256) {
    float acc = 0.0f;
    for (int s = 0; s < S; ++s)
      acc += sc[s] * encF[((size_t)s * B + b) * H2 + d];
    half_t hv = (half_t)acc;
    xinH[b * EH2 + E + d]     = hv;  // GRU input  [e | ctx]
    catH[b * EH3 + E + H + d] = hv;  // cat        [e | h2 | ctx]
  }
}

// ---------------- decoder GRU cell (single block, WMMA) ----------------
__global__ __launch_bounds__(256) void k_dec_gru(
    const half_t* __restrict__ xinH,  // (B, EH2)
    const half_t* __restrict__ Wih,   // (H3, EH2)
    const half_t* __restrict__ Whh,   // (H3, H)
    const float* __restrict__ bih, const float* __restrict__ bhh,
    float* __restrict__ hF, half_t* __restrict__ hH,
    half_t* __restrict__ catH)
{
  const int lane = threadIdx.x & 31;
  const int wave = threadIdx.x >> 5;
  const int r    = lane & 15;
  const int hi   = lane >> 4;
  float hnew[4][8];
  for (int g4 = 0; g4 < 4; ++g4) {
    int g = wave + g4 * 8;
    int mt = g >> 4, j = g & 15;
    const half_t* xrow = xinH + (size_t)mt * 16 * EH2;
    const half_t* hrow = hH + (size_t)mt * 16 * H;
    v8f i0 = {}, i1 = {}, i2 = {}, h0 = {}, h1 = {}, h2 = {};
    i0 = wmma_k_f16(xrow, EH2, Wih + (size_t)(0 * H + j * 16) * EH2, EH2, EH2, i0);
    i1 = wmma_k_f16(xrow, EH2, Wih + (size_t)(1 * H + j * 16) * EH2, EH2, EH2, i1);
    i2 = wmma_k_f16(xrow, EH2, Wih + (size_t)(2 * H + j * 16) * EH2, EH2, EH2, i2);
    h0 = wmma_k_f16(hrow, H, Whh + (size_t)(0 * H + j * 16) * H, H, H, h0);
    h1 = wmma_k_f16(hrow, H, Whh + (size_t)(1 * H + j * 16) * H, H, H, h1);
    h2 = wmma_k_f16(hrow, H, Whh + (size_t)(2 * H + j * 16) * H, H, H, h2);
    int col = j * 16 + r;
    float br = bih[col], bz = bih[H + col], bn = bih[2 * H + col];
    float cr = bhh[col], cz = bhh[H + col], cn = bhh[2 * H + col];
#pragma unroll
    for (int v = 0; v < 8; ++v) {
      int m = mt * 16 + hi * 8 + v;
      float rr = sigf((i0[v] + br) + (h0[v] + cr));
      float zz = sigf((i1[v] + bz) + (h1[v] + cz));
      float nn = tanhf((i2[v] + bn) + rr * (h2[v] + cn));
      float ho = hF[m * H + col];
      hnew[g4][v] = (1.0f - zz) * nn + zz * ho;
    }
  }
  __syncthreads();
  for (int g4 = 0; g4 < 4; ++g4) {
    int g = wave + g4 * 8;
    int mt = g >> 4, j = g & 15;
    int col = j * 16 + r;
#pragma unroll
    for (int v = 0; v < 8; ++v) {
      int m = mt * 16 + hi * 8 + v;
      float hv = hnew[g4][v];
      half_t hh = (half_t)hv;
      hF[m * H + col] = hv;
      hH[m * H + col] = hh;
      catH[m * EH3 + E + col] = hh;  // h2 slot of cat
    }
  }
}

// ---------------- output distribution (block per batch row) ----------------
__global__ __launch_bounds__(256) void k_output(
    const float* __restrict__ logits, const half_t* __restrict__ catH,
    const float* __restrict__ pgenW, const float* __restrict__ pgenB,
    const float* __restrict__ alphas, const int* __restrict__ seq_ext,
    float* __restrict__ outT)
{
  const int b = blockIdx.x;
  const int tid = threadIdx.x;
  __shared__ float red[256];
  // p_gen = sigmoid(cat . pgenW + pgenB)
  float acc = 0.0f;
  for (int i = tid; i < EH3; i += 256) acc += (float)catH[b * EH3 + i] * pgenW[i];
  float pg = block_reduce_sum(acc, red);
  pg = sigf(pg + pgenB[0]);
  // stable softmax over V
  const float* lg = logits + (size_t)b * V;
  float mx = -3.0e38f;
  for (int v = tid; v < V; v += 256) mx = fmaxf(mx, lg[v]);
  mx = block_reduce_max(mx, red);
  float sum = 0.0f;
  for (int v = tid; v < V; v += 256) sum += __expf(lg[v] - mx);
  sum = block_reduce_sum(sum, red);
  float scale = pg / sum;
  float* row = outT + (size_t)b * VE;
  for (int v = tid; v < V; v += 256) row[v] = __expf(lg[v] - mx) * scale + 1e-12f;
  for (int v = V + tid; v < VE; v += 256) row[v] = 1e-12f;
  __threadfence();
  __syncthreads();
  // scatter-add copy distribution (block owns this row)
  float opg = 1.0f - pg;
  for (int s = tid; s < S; s += 256)
    atomicAdd(&row[seq_ext[b * S + s]], opg * alphas[(size_t)s * B + b]);
  __threadfence();
  __syncthreads();
  for (int v = tid; v < VE; v += 256) row[v] = __logf(row[v]);
}

// ---------------- host orchestration ----------------
extern "C" void kernel_launch(void* const* d_in, const int* in_sizes, int n_in,
                              void* d_out, int out_size, void* d_ws, size_t ws_size,
                              hipStream_t stream) {
  (void)in_sizes; (void)n_in; (void)out_size; (void)ws_size;
  const int*   input_seq     = (const int*)  d_in[0];
  const int*   input_seq_ext = (const int*)  d_in[1];
  const int*   target_seq    = (const int*)  d_in[2];
  const float* embedding     = (const float*)d_in[3];
  const float* enc_Wih_f     = (const float*)d_in[4];
  const float* enc_Whh_f     = (const float*)d_in[5];
  const float* enc_bih_f     = (const float*)d_in[6];
  const float* enc_bhh_f     = (const float*)d_in[7];
  const float* enc_Wih_b     = (const float*)d_in[8];
  const float* enc_Whh_b     = (const float*)d_in[9];
  const float* enc_bih_b     = (const float*)d_in[10];
  const float* enc_bhh_b     = (const float*)d_in[11];
  const float* proj_W        = (const float*)d_in[12];
  const float* proj_b        = (const float*)d_in[13];
  const float* bridge_W      = (const float*)d_in[14];
  const float* bridge_b      = (const float*)d_in[15];
  const float* dec_Wih       = (const float*)d_in[16];
  const float* dec_Whh       = (const float*)d_in[17];
  const float* dec_bih       = (const float*)d_in[18];
  const float* dec_bhh       = (const float*)d_in[19];
  const float* attn_Wq       = (const float*)d_in[20];
  const float* attn_We       = (const float*)d_in[21];
  const float* lin_W         = (const float*)d_in[22];
  const float* lin_b         = (const float*)d_in[23];
  const float* pgen_W        = (const float*)d_in[24];
  const float* pgen_b        = (const float*)d_in[25];
  const float* out_W         = (const float*)d_in[26];
  const float* out_b         = (const float*)d_in[27];

  // bump allocator over workspace (256-B aligned blocks)
  char* wsp = (char*)d_ws;
  size_t off = 0;
  auto alloc = [&](size_t elems, size_t esz) -> char* {
    char* p = wsp + off;
    off += (elems * esz + 255) & ~(size_t)255;
    return p;
  };
  half_t* embH     = (half_t*)alloc((size_t)SB * E, 2);
  half_t* WihFh    = (half_t*)alloc((size_t)H3 * E, 2);
  half_t* WhhFh    = (half_t*)alloc((size_t)H3 * H, 2);
  half_t* WihBh    = (half_t*)alloc((size_t)H3 * E, 2);
  half_t* WhhBh    = (half_t*)alloc((size_t)H3 * H, 2);
  half_t* projWh   = (half_t*)alloc((size_t)H * H2, 2);
  half_t* bridgeWh = (half_t*)alloc((size_t)H * H2, 2);
  half_t* decWihh  = (half_t*)alloc((size_t)H3 * EH2, 2);
  half_t* decWhhh  = (half_t*)alloc((size_t)H3 * H, 2);
  half_t* Wqh      = (half_t*)alloc((size_t)H * H, 2);
  half_t* linWh    = (half_t*)alloc((size_t)H * EH3, 2);
  half_t* outWh    = (half_t*)alloc((size_t)V * H, 2);
  half_t* encH     = (half_t*)alloc((size_t)SB * H2, 2);
  float*  encF     = (float*) alloc((size_t)SB * H2, 4);
  float*  projF    = (float*) alloc((size_t)SB * H, 4);
  half_t* hTh      = (half_t*)alloc((size_t)B * H2, 2);
  float*  hdecF    = (float*) alloc((size_t)B * H, 4);
  half_t* hdecH    = (half_t*)alloc((size_t)B * H, 2);
  float*  qF       = (float*) alloc((size_t)B * H, 4);
  float*  alphas   = (float*) alloc((size_t)S * B, 4);
  half_t* xinH     = (half_t*)alloc((size_t)B * EH2, 2);
  half_t* catH     = (half_t*)alloc((size_t)B * EH3, 2);
  float*  linF     = (float*) alloc((size_t)B * H, 4);
  half_t* linH     = (half_t*)alloc((size_t)B * H, 2);
  float*  logits   = (float*) alloc((size_t)B * V, 4);

  auto cast16 = [&](const float* src, half_t* dst, int n) {
    k_cast_f32_f16<<<(n + 255) / 256, 256, 0, stream>>>(src, dst, n);
  };
  auto gemm = [&](const half_t* A, const half_t* W, const float* bias,
                  float* C, half_t* C16, int M, int N, int K, int act) {
    int tiles = (M / 16) * (N / 16);
    int blocks = (tiles * 32 + 255) / 256;
    k_gemm<<<blocks, 256, 0, stream>>>(A, W, bias, C, C16, M, N, K, act);
  };

  // weight casts (f32 -> f16 WMMA operands)
  cast16(enc_Wih_f, WihFh, H3 * E);
  cast16(enc_Whh_f, WhhFh, H3 * H);
  cast16(enc_Wih_b, WihBh, H3 * E);
  cast16(enc_Whh_b, WhhBh, H3 * H);
  cast16(proj_W, projWh, H * H2);
  cast16(bridge_W, bridgeWh, H * H2);
  cast16(dec_Wih, decWihh, H3 * EH2);
  cast16(dec_Whh, decWhhh, H3 * H);
  cast16(attn_Wq, Wqh, H * H);
  cast16(lin_W, linWh, H * EH3);
  cast16(out_W, outWh, V * H);

  // encoder
  k_embed_src<<<(SB * E + 255) / 256, 256, 0, stream>>>(input_seq, embedding, embH);
  k_encoder<<<2, 256, 0, stream>>>(embH, WihFh, WhhFh, enc_bih_f, enc_bhh_f,
                                   WihBh, WhhBh, enc_bih_b, enc_bhh_b,
                                   encH, encF, hTh);
  // proj = enc_states @ proj_W.T + proj_b   (12800 x 256, K=512)
  gemm(encH, projWh, proj_b, projF, nullptr, SB, H, H2, 0);
  // h0 = tanh([hf|hb] @ bridge_W.T + bridge_b)
  gemm(hTh, bridgeWh, bridge_b, hdecF, hdecH, B, H, H2, 1);
  // output row t=0 is zeros
  k_zero<<<(B * VE + 255) / 256, 256, 0, stream>>>((float*)d_out, B * VE);

  // decoder (teacher forcing), t = 1..T-1
  for (int t = 1; t < T; ++t) {
    k_embed_tok<<<(B * E + 255) / 256, 256, 0, stream>>>(target_seq, embedding,
                                                         xinH, catH, t - 1);
    // q = h @ Wq.T
    gemm(hdecH, Wqh, nullptr, qF, nullptr, B, H, H, 0);
    k_attn<<<B, 256, 0, stream>>>(qF, projF, attn_We, encF, alphas, xinH, catH);
    k_dec_gru<<<1, 256, 0, stream>>>(xinH, decWihh, decWhhh, dec_bih, dec_bhh,
                                     hdecF, hdecH, catH);
    // lin = cat @ lin_W.T + lin_b   (32 x 256, K=896)
    gemm(catH, linWh, lin_b, linF, linH, B, H, EH3, 0);
    // logits = lin @ out_W.T + out_b   (32 x 50000, K=256) -- out_W stays in L2
    gemm(linH, outWh, out_b, logits, nullptr, B, V, H, 0);
    k_output<<<B, 256, 0, stream>>>(logits, catH, pgen_W, pgen_b, alphas,
                                    input_seq_ext,
                                    (float*)d_out + (size_t)t * B * VE);
  }
}
